// MultiHeadAttention_24378234372210
// MI455X (gfx1250) — matrix operand order repaired and checked by reading, not compiled, not run on hardware
//
#include <hip/hip_runtime.h>

typedef __attribute__((ext_vector_type(16))) _Float16 v16h;
typedef __attribute__((ext_vector_type(8)))  float    v8f;

#define DM   512
#define SEQ  256
#define NB   2
#define NH   8
#define DKV  64

__device__ __forceinline__ v8f wmma16(v16h a, v16h b, v8f c) {
  // D = A(16x32 f16) * B(32x16 f16) + C(16x16 f32)
  return __builtin_amdgcn_wmma_f32_16x16x32_f16(false, a, false, b, (short)0, c,
                                                false, false);
}

// Resolve con-vec row (b-relative float offset into qs/ks) for (h, s2, c).
// Derived from the triu/flip/pad + reinterpreting-reshape algebra of get_con_vecs.
__device__ __forceinline__ int convec_off(int h, int s2, int c, bool& valid) {
  int r = s2 & 7;
  int i = h * 32 + (s2 >> 3);
  int u = (r & 3) * 64 + 30 + (c >> 3);
  valid = (r < 4) ? (u + i >= 256) : (u + i <= 255);
  int hsrc = u >> 5;
  int stok = ((u & 31) << 3) + (c & 7);
  return stok * DM + hsrc * DKV;
}

// ---------------- Kernel 1: Q/K projections (GEMM 512x512x512 each) ----------
__global__ void __launch_bounds__(256) proj_kernel(
    const float* __restrict__ Q, const float* __restrict__ K,
    const float* __restrict__ WQ, const float* __restrict__ bQ,
    const float* __restrict__ WK, const float* __restrict__ bK,
    float* __restrict__ qs, float* __restrict__ ks) {
  int lane = threadIdx.x & 31;
  int w    = threadIdx.x >> 5;
  int wid  = blockIdx.x * 8 + w;        // 0..2047 wave tasks
  int which = wid >> 10;                // 0 = Q, 1 = K
  int rem   = wid & 1023;
  int t0 = (rem >> 5) * 16;             // token tile (512 tokens)
  int n0 = (rem & 31) * 16;             // output-col tile
  const float* X    = which ? K  : Q;
  const float* W    = which ? WK : WQ;
  const float* bias = which ? bK : bQ;
  float*       dst  = which ? ks : qs;

  int  lo = lane & 15;
  bool hi = lane >= 16;
  int  ab = hi ? 8 : 0;
  int  kb = hi ? 16 : 0;
  const float* xr = X + (t0 + lo) * DM;
  int col = n0 + lo;

  v8f acc = {};
  for (int ch = 0; ch < DM; ch += 32) {
    v16h a, bf;
#pragma unroll
    for (int t = 0; t < 8; ++t) {
      a[t]     = (_Float16)xr[ch + kb + t];
      a[t + 8] = (_Float16)xr[ch + kb + 8 + t];
    }
#pragma unroll
    for (int t = 0; t < 16; ++t)
      bf[t] = (_Float16)W[(ch + kb + t) * DM + col];
    acc = wmma16(a, bf, acc);
  }
  float bv = bias[col];
#pragma unroll
  for (int r = 0; r < 8; ++r) {
    int row = t0 + r + (hi ? 8 : 0);
    dst[row * DM + col] = acc[r] + bv;
  }
}

// ---------------- Kernel 2: materialize Vc(=Kc) as f16 [b][h][k][d][c] -------
__global__ void __launch_bounds__(256) vcb_kernel(
    const float* __restrict__ ks, _Float16* __restrict__ vcb) {
  int idx = blockIdx.x * 256 + threadIdx.x;   // 131072 = b*h*k*c
  int c = idx & 31;
  int k = (idx >> 5) & 255;
  int h = (idx >> 13) & 7;
  int b = idx >> 16;
  bool v; int off = convec_off(h, k, c, v);
  const float* src = ks + b * (SEQ * DM) + off;
  _Float16* dst = vcb + ((((b * NH + h) * SEQ + k) * DKV) << 5) + c;
#pragma unroll 8
  for (int d = 0; d < 64; ++d)
    dst[d * 32] = v ? (_Float16)src[d] : (_Float16)0.f;
}

// ---------------- Kernel 3: fused scores -> softmax(c) -> context ------------
__global__ void __launch_bounds__(128) attn_kernel(
    const float* __restrict__ qs, const float* __restrict__ ks,
    const _Float16* __restrict__ vcb, const unsigned char* __restrict__ mask,
    float* __restrict__ ctx, float* __restrict__ attn_out) {
  extern __shared__ char smem[];   // 4 waves * 48KB (32KB scores f32 + 16KB attn f16)
  int  lane = threadIdx.x & 31;
  int  w    = threadIdx.x >> 5;
  int  lo   = lane & 15;
  bool hi   = lane >= 16;
  int  qt = blockIdx.x & 15;
  int  h  = (blockIdx.x >> 4) & 7;
  int  b  = blockIdx.x >> 7;
  int  q0 = qt * 16;

  float*    sc = (float*)(smem + w * 49152);             // [16][512] f32, j=k*32+c
  _Float16* at = (_Float16*)(smem + w * 49152 + 32768);  // [16][512] f16

  const float* qsb = qs + b * (SEQ * DM);
  const float* ksb = ks + b * (SEQ * DM);

  v8f zero = {};
  v8f cacc0 = zero, cacc1 = zero, cacc2 = zero, cacc3 = zero;

  for (int j = 0; j < 4; ++j) {
    int kt = w + 4 * j;
    int k0 = kt * 16;
    // ---- phase 1: scores[q][k][c] = (Qc . Kc)/8 via WMMA ----
    for (int c = 0; c < 32; ++c) {
      bool vq; int offq = convec_off(h, q0 + lo, c, vq);
      bool vk; int offk = convec_off(h, k0 + lo, c, vk);
      const float* srq = qsb + offq;
      const float* srk = ksb + offk;
      int ab = hi ? 8 : 0;
      int kb = hi ? 16 : 0;
      v16h a0, a1, b0, b1;
#pragma unroll
      for (int t = 0; t < 8; ++t) {
        a0[t]     = vq ? (_Float16)srq[kb + t]           : (_Float16)0.f;
        a0[t + 8] = vq ? (_Float16)srq[kb + 8 + t]       : (_Float16)0.f;
        a1[t]     = vq ? (_Float16)srq[32 + kb + t]      : (_Float16)0.f;
        a1[t + 8] = vq ? (_Float16)srq[32 + kb + 8 + t]  : (_Float16)0.f;
      }
#pragma unroll
      for (int t = 0; t < 16; ++t) {
        b0[t] = vk ? (_Float16)srk[kb + t]      : (_Float16)0.f;
        b1[t] = vk ? (_Float16)srk[32 + kb + t] : (_Float16)0.f;
      }
      v8f acc = zero;
      acc = wmma16(a0, b0, acc);
      acc = wmma16(a1, b1, acc);
#pragma unroll
      for (int r = 0; r < 8; ++r) {
        int ql = r + (hi ? 8 : 0);
        sc[ql * 512 + lo * 32 + c] = acc[r] * 0.125f;
      }
    }
    // ---- phase 2: softmax over c per (q,k); masked -> uniform 1/32 ----
    for (int t = 0; t < 8; ++t) {
      int p  = lane + 32 * t;
      int ql = p >> 4, kl = p & 15;
      bool m = mask[(b * SEQ + q0 + ql) * SEQ + k0 + kl] != 0;
      float v[32];
      if (m) {
#pragma unroll
        for (int c = 0; c < 32; ++c) v[c] = 0.03125f;
      } else {
        float mx = -1e30f;
#pragma unroll
        for (int c = 0; c < 32; ++c) {
          v[c] = sc[ql * 512 + kl * 32 + c];
          mx = fmaxf(mx, v[c]);
        }
        float s = 0.f;
#pragma unroll
        for (int c = 0; c < 32; ++c) { v[c] = expf(v[c] - mx); s += v[c]; }
        float inv = 1.f / s;
#pragma unroll
        for (int c = 0; c < 32; ++c) v[c] *= inv;
      }
#pragma unroll
      for (int c = 0; c < 32; ++c)
        at[ql * 512 + kl * 32 + c] = (_Float16)v[c];
      // softmax over c sums to 1 -> attn_out is identically 1.0
      attn_out[((b * NH + h) * SEQ + q0 + ql) * SEQ + k0 + kl] = 1.0f;
    }
    // ---- phase 3: ctx += attn(16x512) @ Vc(512x64), chunked K=32 ----
    for (int jc = 0; jc < 16; ++jc) {
      v16h a;
      int abase = jc * 32 + (hi ? 16 : 0);
#pragma unroll
      for (int t = 0; t < 8; ++t) {
        a[t]     = at[lo * 512 + abase + t];
        a[t + 8] = at[lo * 512 + abase + 8 + t];
      }
      const _Float16* vb = vcb + ((((b * NH + h) * SEQ + k0 + jc) * DKV) << 5);
      int cb = hi ? 16 : 0;
      {
        const _Float16* p0 = vb + (0 + lo) * 32 + cb;
        v16h bf;
#pragma unroll
        for (int t = 0; t < 16; ++t) bf[t] = p0[t];
        cacc0 = wmma16(a, bf, cacc0);
      }
      {
        const _Float16* p0 = vb + (16 + lo) * 32 + cb;
        v16h bf;
#pragma unroll
        for (int t = 0; t < 16; ++t) bf[t] = p0[t];
        cacc1 = wmma16(a, bf, cacc1);
      }
      {
        const _Float16* p0 = vb + (32 + lo) * 32 + cb;
        v16h bf;
#pragma unroll
        for (int t = 0; t < 16; ++t) bf[t] = p0[t];
        cacc2 = wmma16(a, bf, cacc2);
      }
      {
        const _Float16* p0 = vb + (48 + lo) * 32 + cb;
        v16h bf;
#pragma unroll
        for (int t = 0; t < 16; ++t) bf[t] = p0[t];
        cacc3 = wmma16(a, bf, cacc3);
      }
    }
  }
  // ---- cross-wave reduction of context partials ----
  __syncthreads();
  float* red = (float*)smem;   // [4][16][64]
#pragma unroll
  for (int r = 0; r < 8; ++r) {
    int ql = r + (hi ? 8 : 0);
    red[(w * 16 + ql) * 64 + 0  + lo] = cacc0[r];
    red[(w * 16 + ql) * 64 + 16 + lo] = cacc1[r];
    red[(w * 16 + ql) * 64 + 32 + lo] = cacc2[r];
    red[(w * 16 + ql) * 64 + 48 + lo] = cacc3[r];
  }
  __syncthreads();
  for (int e = threadIdx.x; e < 1024; e += 128) {
    float s = red[e] + red[1024 + e] + red[2048 + e] + red[3072 + e];
    int ql = e >> 6, d = e & 63;
    ctx[(b * SEQ + q0 + ql) * DM + h * DKV + d] = s;
  }
}

// ---------------- Kernel 4: out = ctx@Wo + bo + Q, then LayerNorm ------------
__global__ void __launch_bounds__(256) out_kernel(
    const float* __restrict__ ctx, const float* __restrict__ Wo,
    const float* __restrict__ bo, const float* __restrict__ Q,
    const float* __restrict__ gamma, const float* __restrict__ beta,
    float* __restrict__ out) {
  __shared__ float ybuf[16 * 512];
  __shared__ float redS[16][16];
  __shared__ float redQ[16][16];
  __shared__ float stats[16][2];
  int  lane = threadIdx.x & 31;
  int  w    = threadIdx.x >> 5;
  int  lo   = lane & 15;
  bool hi   = lane >= 16;
  int  t0   = blockIdx.x * 16;
  const float* xr = ctx + (t0 + lo) * DM;
  int ab = hi ? 8 : 0;
  int kb = hi ? 16 : 0;

  v8f zero = {};
  v8f acc[4] = {zero, zero, zero, zero};
  for (int ch = 0; ch < DM; ch += 32) {
    v16h a;
#pragma unroll
    for (int t = 0; t < 8; ++t) {
      a[t]     = (_Float16)xr[ch + kb + t];
      a[t + 8] = (_Float16)xr[ch + kb + 8 + t];
    }
#pragma unroll
    for (int j = 0; j < 4; ++j) {
      int col = (w + 8 * j) * 16 + lo;
      v16h bf;
#pragma unroll
      for (int t = 0; t < 16; ++t)
        bf[t] = (_Float16)Wo[(ch + kb + t) * DM + col];
      acc[j] = wmma16(a, bf, acc[j]);
    }
  }
#pragma unroll
  for (int j = 0; j < 4; ++j) {
    int col = (w + 8 * j) * 16 + lo;
#pragma unroll
    for (int r = 0; r < 8; ++r) {
      int lr = r + (hi ? 8 : 0);
      ybuf[lr * 512 + col] = acc[j][r] + bo[col] + Q[(t0 + lr) * DM + col];
    }
  }
  __syncthreads();
  int lrow = threadIdx.x >> 4;
  int seg  = threadIdx.x & 15;
  float s1 = 0.f, s2 = 0.f;
#pragma unroll
  for (int e = 0; e < 32; ++e) {
    float x = ybuf[lrow * 512 + seg * 32 + e];
    s1 += x; s2 += x * x;
  }
  redS[lrow][seg] = s1;
  redQ[lrow][seg] = s2;
  __syncthreads();
  if (seg == 0) {
    float a = 0.f, q2 = 0.f;
#pragma unroll
    for (int t = 0; t < 16; ++t) { a += redS[lrow][t]; q2 += redQ[lrow][t]; }
    float mu  = a * (1.f / 512.f);
    float var = q2 * (1.f / 512.f) - mu * mu;
    stats[lrow][0] = mu;
    stats[lrow][1] = rsqrtf(var + 1e-5f);
  }
  __syncthreads();
  float mu = stats[lrow][0], rs = stats[lrow][1];
  for (int e = 0; e < 32; ++e) {
    int col = seg * 32 + e;
    float x = ybuf[lrow * 512 + col];
    out[(t0 + lrow) * DM + col] = (x - mu) * rs * gamma[col] + beta[col];
  }
}

extern "C" void kernel_launch(void* const* d_in, const int* in_sizes, int n_in,
                              void* d_out, int out_size, void* d_ws, size_t ws_size,
                              hipStream_t stream) {
  const float* Q  = (const float*)d_in[0];
  const float* K  = (const float*)d_in[1];
  const unsigned char* mask = (const unsigned char*)d_in[3];  // jax bool = 1 byte
  const float* WQ = (const float*)d_in[4];
  const float* bQ = (const float*)d_in[5];
  const float* WK = (const float*)d_in[6];
  const float* bK = (const float*)d_in[7];
  const float* Wo = (const float*)d_in[10];
  const float* bo = (const float*)d_in[11];
  const float* gamma = (const float*)d_in[12];
  const float* beta  = (const float*)d_in[13];

  float* y_out    = (float*)d_out;                  // (2,256,512)
  float* attn_out = y_out + NB * SEQ * DM;          // (2,8,256,256), all 1.0

  float* qs  = (float*)d_ws;                        // 262144 f32
  float* ks  = qs  + NB * SEQ * DM;                 // 262144 f32
  float* ctx = ks  + NB * SEQ * DM;                 // 262144 f32
  _Float16* vcb = (_Float16*)(ctx + NB * SEQ * DM); // 8388608 f16 (16 MB)

  proj_kernel<<<256, 256, 0, stream>>>(Q, K, WQ, bQ, WK, bK, qs, ks);
  vcb_kernel<<<512, 256, 0, stream>>>(ks, vcb);
  attn_kernel<<<256, 128, 192 * 1024, stream>>>(qs, ks, vcb, mask, ctx, attn_out);
  out_kernel<<<32, 256, 0, stream>>>(ctx, Wo, bo, Q, gamma, beta, y_out);
}
